// BiAttention_78683800863015
// MI455X (gfx1250) — matrix operand order repaired and checked by reading, not compiled, not run on hardware
//
#include <hip/hip_runtime.h>
#include <hip/hip_bf16.h>

// ---------------------------------------------------------------------------
// BiAttention on MI455X (gfx1250). All GEMMs on v_wmma_f32_16x16x32_bf16;
// the two dominant GEMMs stage their shared weight slabs into LDS via the
// Tensor Data Mover (tensor_load_to_lds, double-buffered, s_wait_tensorcnt),
// fp32 score matrix never materialized, softmax normalization folded into the
// GEMM epilogues. Deterministic (atomic-max only; adds are tree reductions).
// ---------------------------------------------------------------------------

typedef __bf16 bf16;
typedef __attribute__((ext_vector_type(16))) __bf16 v16bf;
typedef __attribute__((ext_vector_type(8)))  __bf16 v8bf;
typedef __attribute__((ext_vector_type(8)))  float  v8f;
typedef __attribute__((ext_vector_type(4)))  unsigned int u32x4;
typedef __attribute__((ext_vector_type(4)))  int  i32x4;
typedef __attribute__((ext_vector_type(8)))  int  i32x8;

static constexpr int BSZ = 8;
static constexpr int LSEQ = 2048;
static constexpr int DMODEL = 768;
static constexpr int HDIM = 64;

#if defined(__has_builtin)
#if __has_builtin(__builtin_amdgcn_tensor_load_to_lds) && \
    __has_builtin(__builtin_amdgcn_s_wait_tensorcnt)
#define USE_TDM 1
#endif
#endif

// ---- fragment helpers ------------------------------------------------------
// A-matrix 16x32 bf16 (row-major source, leading dim lda):
//   lanes 0-15 : M=lane,    elems 0-15 -> K=0..15
//   lanes 16-31: M=lane-16, elems 0-15 -> K=16..31
__device__ __forceinline__ v16bf load_a_frag(const bf16* base, int lda, int lane) {
    const int m = lane & 15;
    const int k = (lane >> 4) * 16;
    const bf16* p = base + (size_t)m * lda + k;
    v8bf lo = *(const v8bf*)(p);
    v8bf hi = *(const v8bf*)(p + 8);
    return __builtin_shufflevector(lo, hi, 0, 1, 2, 3, 4, 5, 6, 7,
                                           8, 9, 10, 11, 12, 13, 14, 15);
}

// Same fragment but sourced from fp32 (two v8f loads + pack-convert).
__device__ __forceinline__ v16bf load_a_frag_f32(const float* __restrict__ base,
                                                 int lda, int lane) {
    const int m = lane & 15;
    const int k = (lane >> 4) * 16;
    const float* q = base + (size_t)m * lda + k;
    v8f lo = *(const v8f*)(q);
    v8f hi = *(const v8f*)(q + 8);
    v16bf a;
#pragma unroll
    for (int e = 0; e < 8; ++e) { a[e] = (bf16)lo[e]; a[e + 8] = (bf16)hi[e]; }
    return a;
}

__device__ __forceinline__ v8f wmma_bf16(v16bf a, v16bf b, v8f c) {
    return __builtin_amdgcn_wmma_f32_16x16x32_bf16(
        false, a, false, b, (short)0, c, false, false);
}

// Deterministic float atomic max via integer atomics (order-independent).
__device__ __forceinline__ void atomicMaxF(float* addr, float val) {
    if (!(val < 0.0f)) atomicMax((int*)addr, __float_as_int(val));
    else               atomicMin((unsigned int*)addr, __float_as_uint(val));
}

// ---- TDM: async-load a 16 x BK bf16 tile (row stride LSEQ) into LDS -------
static constexpr int BK = 256;                 // K-slab per LDS buffer
static constexpr int SLAB_BYTES = 16 * BK * (int)sizeof(bf16);   // 8 KB

#ifdef USE_TDM
__device__ __forceinline__ void tdm_load_tile(unsigned lds_off_bytes,
                                              const bf16* gptr) {
    unsigned long long ga = (unsigned long long)gptr;
    u32x4 g0;
    g0[0] = 1u;                                    // count=1, user descriptor
    g0[1] = lds_off_bytes;                         // lds_addr (bytes)
    g0[2] = (unsigned)(ga & 0xFFFFFFFFu);          // global_addr[31:0]
    g0[3] = (unsigned)((ga >> 32) & 0x1FFFFFFu)    // global_addr[56:32]
          | (2u << 30);                            // type = 2 ("image")
    i32x8 g1;
    g1[0] = 0x10000;                               // data_size=2B, mask=0
    g1[1] = (int)((LSEQ & 0xFFFF) << 16);          // tensor_dim0 lo16
    g1[2] = (int)(((LSEQ >> 16) & 0xFFFF)          // tensor_dim0 hi16
          | ((LSEQ & 0xFFFF) << 16));              // tensor_dim1 lo16
    g1[3] = (int)(((LSEQ >> 16) & 0xFFFF)          // tensor_dim1 hi16
          | (BK << 16));                           // tile_dim0 = BK elems
    g1[4] = 16;                                    // tile_dim1 = 16 rows
    g1[5] = LSEQ;                                  // tensor_dim0_stride lo32
    g1[6] = 0;
    g1[7] = 0;
    i32x4 z = {0, 0, 0, 0};
#if __clang_major__ >= 23
    i32x8 z8 = {0, 0, 0, 0, 0, 0, 0, 0};
    __builtin_amdgcn_tensor_load_to_lds(g0, g1, z, z, z8, 0);
#else
    __builtin_amdgcn_tensor_load_to_lds(g0, g1, z, z, 0);
#endif
}
#endif

// ---- stage 1: transpose-cast  X[b,s,d] f32 -> Xt[b,d,s] bf16 --------------
__global__ void __launch_bounds__(256) transpose_cast_kernel(
    const float* __restrict__ X, bf16* __restrict__ Xt) {
    __shared__ bf16 tile[32][33];
    const int b  = blockIdx.z;
    const int d0 = blockIdx.x * 32;
    const int s0 = blockIdx.y * 32;
    const int tx = threadIdx.x & 31;
    const int ty = threadIdx.x >> 5;            // 0..7
    const float* Xb = X + (size_t)b * LSEQ * DMODEL;
    bf16* Xtb = Xt + (size_t)b * DMODEL * LSEQ;
#pragma unroll
    for (int j = 0; j < 32; j += 8)
        tile[ty + j][tx] = (bf16)Xb[(size_t)(s0 + ty + j) * DMODEL + d0 + tx];
    __syncthreads();
#pragma unroll
    for (int j = 0; j < 32; j += 8)
        Xtb[(size_t)(d0 + ty + j) * LSEQ + s0 + tx] = tile[tx][ty + j];
}

// ---- stage 2: W[d,h] f32 -> Wt[h,d] bf16 (tiny) ---------------------------
__global__ void __launch_bounds__(256) wcast_kernel(
    const float* __restrict__ W, bf16* __restrict__ Wt) {
    int i = blockIdx.x * 256 + threadIdx.x;     // over H*D
    if (i < HDIM * DMODEL) {
        int h = i / DMODEL, d = i - h * DMODEL;
        Wt[i] = (bf16)W[(size_t)d * HDIM + h];
    }
}

// ---- stage 3: projection  P[b,s,:] = X[b,s,:] @ W + bias ------------------
__global__ void __launch_bounds__(128) proj_kernel(
    const float* __restrict__ X, const bf16* __restrict__ Wt,
    const float* __restrict__ bias, bf16* __restrict__ P) {
    const int lane = threadIdx.x & 31;
    const int wave = threadIdx.x >> 5;
    const int m0 = blockIdx.x * 16;
    const int b  = blockIdx.z;
    const int n0 = wave * 16;
    const float* Xb = X + ((size_t)b * LSEQ + m0) * DMODEL;
    const int n  = lane & 15;
    const int kb = (lane >> 4) * 16;
    v8f acc = {};
    for (int k0 = 0; k0 < DMODEL; k0 += 32) {
        v16bf a  = load_a_frag_f32(Xb + k0, DMODEL, lane);
        v16bf bf_ = *(const v16bf*)(Wt + (size_t)(n0 + n) * DMODEL + k0 + kb);
        acc = wmma_bf16(a, bf_, acc);
    }
    const int mo = (lane >> 4) * 8;
    const float bv = bias[n0 + n];
#pragma unroll
    for (int i = 0; i < 8; ++i)
        P[((size_t)b * LSEQ + m0 + mo + i) * HDIM + (n0 + n)] = (bf16)(acc[i] + bv);
}

// ---- shared score-tile GEMM (K = HDIM = 64) -------------------------------
__device__ __forceinline__ v8f score_tile(const bf16* __restrict__ Aprow,
                                          const bf16* __restrict__ Bprow,
                                          int lane) {
    const int n  = lane & 15;
    const int kb = (lane >> 4) * 16;
    v8f acc = {};
#pragma unroll
    for (int k0 = 0; k0 < HDIM; k0 += 32) {
        v16bf a  = load_a_frag(Aprow + k0, HDIM, lane);
        v16bf bf_ = *(const v16bf*)(Bprow + (size_t)n * HDIM + k0 + kb);
        acc = wmma_bf16(a, bf_, acc);
    }
    return acc;
}

// ---- stage 4a: init row/col maxima to -inf --------------------------------
__global__ void __launch_bounds__(256) init_neg_inf_kernel(float* p, int n) {
    int i = blockIdx.x * 256 + threadIdx.x;
    if (i < n) p[i] = -__builtin_inff();
}

// ---- stage 4b: score pass 1 -> row/col maxima -----------------------------
__global__ void __launch_bounds__(256) scores_max_kernel(
    const bf16* __restrict__ Ap, const bf16* __restrict__ Bp,
    float* __restrict__ rmax, float* __restrict__ cmax) {
    const int lane = threadIdx.x & 31;
    const int wave = threadIdx.x >> 5;
    const int b  = blockIdx.z;
    const int s0 = blockIdx.y * 16;
    const int t0 = blockIdx.x * 128 + wave * 16;
    v8f acc = score_tile(Ap + ((size_t)b * LSEQ + s0) * HDIM,
                         Bp + ((size_t)b * LSEQ + t0) * HDIM, lane);
    const int n  = lane & 15;
    const int hf = lane >> 4;
    const int mo = hf * 8;
    float cm = acc[0];
#pragma unroll
    for (int i = 1; i < 8; ++i) cm = fmaxf(cm, acc[i]);
    cm = fmaxf(cm, __shfl_xor(cm, 16, 32));
    if (hf == 0) atomicMaxF(&cmax[(size_t)b * LSEQ + t0 + n], cm);
#pragma unroll
    for (int i = 0; i < 8; ++i) {
        float rm = acc[i];
        rm = fmaxf(rm, __shfl_xor(rm, 1, 32));
        rm = fmaxf(rm, __shfl_xor(rm, 2, 32));
        rm = fmaxf(rm, __shfl_xor(rm, 4, 32));
        rm = fmaxf(rm, __shfl_xor(rm, 8, 32));
        if (n == 0) atomicMaxF(&rmax[(size_t)b * LSEQ + s0 + mo + i], rm);
    }
}

// ---- stage 5: score pass 2 -> bf16 exp-weights, both orientations ---------
__global__ void __launch_bounds__(256) scores_weights_kernel(
    const bf16* __restrict__ Ap, const bf16* __restrict__ Bp,
    const float* __restrict__ rmax, const float* __restrict__ cmax,
    bf16* __restrict__ WA, bf16* __restrict__ WBt) {
    const int lane = threadIdx.x & 31;
    const int wave = threadIdx.x >> 5;
    const int b  = blockIdx.z;
    const int s0 = blockIdx.y * 16;
    const int t0 = blockIdx.x * 128 + wave * 16;
    v8f acc = score_tile(Ap + ((size_t)b * LSEQ + s0) * HDIM,
                         Bp + ((size_t)b * LSEQ + t0) * HDIM, lane);
    const int n  = lane & 15;
    const int mo = (lane >> 4) * 8;
    const float cmv = cmax[(size_t)b * LSEQ + t0 + n];
    bf16* WAb  = WA  + (size_t)b * LSEQ * LSEQ;
    bf16* WBtb = WBt + (size_t)b * LSEQ * LSEQ;
    v8bf wbv;
#pragma unroll
    for (int i = 0; i < 8; ++i) {
        float x   = acc[i];
        float rmv = rmax[(size_t)b * LSEQ + s0 + mo + i];
        WAb[(size_t)(s0 + mo + i) * LSEQ + t0 + n] = (bf16)__expf(x - rmv);
        wbv[i] = (bf16)__expf(x - cmv);
    }
    *(v8bf*)(WBtb + (size_t)(t0 + n) * LSEQ + s0 + mo) = wbv;
}

// ---- stage 6: exact softmax denominators (contiguous row sums) ------------
__global__ void __launch_bounds__(256) rowsum_kernel(
    const bf16* __restrict__ M, float* __restrict__ out) {
    __shared__ float red[256];
    const size_t r = (size_t)blockIdx.z * LSEQ + blockIdx.x;
    const bf16* row = M + r * LSEQ;
    float s = 0.f;
    for (int j = threadIdx.x; j < LSEQ; j += 256) s += (float)row[j];
    red[threadIdx.x] = s;
    __syncthreads();
    for (int off = 128; off > 0; off >>= 1) {
        if ((int)threadIdx.x < off) red[threadIdx.x] += red[threadIdx.x + off];
        __syncthreads();
    }
    if (threadIdx.x == 0) out[r] = red[0];
}

// ---- stage 7: Out[b,r,d] = (Wm[b,r,:] @ Vt[b,d,:]^T) / sum[b,r] -----------
// TDM double-buffers the 16 x BK weight slab (shared by all 4 waves) into
// LDS; each wave holds two 16x16 accumulators (d and d+64), so each LDS
// A-fragment pair feeds two v_wmma.
__global__ void __launch_bounds__(128) weighted_sum_kernel(
    const bf16* __restrict__ Wm,   // [B, L, L] row-major (row=out row, col=K)
    const bf16* __restrict__ Vt,   // [B, D, L] (row=d, col=K)
    const float* __restrict__ dsum,// [B, L]
    float* __restrict__ Out) {     // [B, L, D]
    __shared__ bf16 smem[2][16 * BK];
    const int lane = threadIdx.x & 31;
    const int wave = threadIdx.x >> 5;
    const int b  = blockIdx.z;
    const int r0 = blockIdx.y * 16;
    const int d0 = blockIdx.x * 128 + wave * 16;
    const int n  = lane & 15;
    const int kb = (lane >> 4) * 16;
    const int am = lane & 15;
    const int ak = (lane >> 4) * 16;
    const bf16* Wrow = Wm + ((size_t)b * LSEQ + r0) * LSEQ;
    const bf16* V0 = Vt + ((size_t)b * DMODEL + d0 + n) * LSEQ;
    const bf16* V1 = V0 + (size_t)64 * LSEQ;
    v8f acc0 = {}, acc1 = {};

    // Escape the shared buffer's address: the TDM engine writes it behind the
    // compiler's back, so alias analysis must not fold the LDS loads to undef.
    {
        const bf16* esc = &smem[0][0];
        asm volatile("" ::"v"(esc) : "memory");
    }

#ifdef USE_TDM
    if (wave == 0) {
        tdm_load_tile(0u, Wrow);
        __builtin_amdgcn_s_wait_tensorcnt(0);
    }
    __syncthreads();
#else
    // fallback: cooperative copy of slab 0
    for (int j = threadIdx.x; j < 16 * BK / 8; j += 128)
        *(v8bf*)&smem[0][j * 8] = *(const v8bf*)(Wrow + (j / (BK / 8)) * LSEQ +
                                                 (j % (BK / 8)) * 8);
    __syncthreads();
#endif

    int buf = 0;
    for (int s = 0; s < LSEQ / BK; ++s) {
#ifdef USE_TDM
        if (wave == 0 && s + 1 < LSEQ / BK)
            tdm_load_tile((unsigned)((buf ^ 1) * SLAB_BYTES), Wrow + (s + 1) * BK);
#endif
        // Opaque potential writer once per slab: forces re-load of the LDS
        // slab written by the TDM, while still letting the compiler batch the
        // 16 ds_load_b128 within the slab.
        asm volatile("" ::: "memory");
        const bf16* As = &smem[buf][0];
        const int kg = s * BK;
#pragma unroll
        for (int kk = 0; kk < BK; kk += 32) {
            const bf16* p = As + am * BK + kk + ak;
            v8bf lo = *(const v8bf*)(p);
            v8bf hi = *(const v8bf*)(p + 8);
            v16bf a = __builtin_shufflevector(lo, hi, 0, 1, 2, 3, 4, 5, 6, 7,
                                                      8, 9, 10, 11, 12, 13, 14, 15);
            v16bf b0 = *(const v16bf*)(V0 + kg + kk + kb);
            v16bf b1 = *(const v16bf*)(V1 + kg + kk + kb);
            acc0 = wmma_bf16(a, b0, acc0);
            acc1 = wmma_bf16(a, b1, acc1);
        }
#ifdef USE_TDM
        if (wave == 0) __builtin_amdgcn_s_wait_tensorcnt(0);
        __syncthreads();
#else
        __syncthreads();
        if (s + 1 < LSEQ / BK) {
            const bf16* src = Wrow + (s + 1) * BK;
            for (int j = threadIdx.x; j < 16 * BK / 8; j += 128)
                *(v8bf*)&smem[buf ^ 1][j * 8] =
                    *(const v8bf*)(src + (j / (BK / 8)) * LSEQ + (j % (BK / 8)) * 8);
            __syncthreads();
        }
#endif
        buf ^= 1;
    }

    const int mo = (lane >> 4) * 8;
#pragma unroll
    for (int i = 0; i < 8; ++i) {
        int m = r0 + mo + i;
        float inv = 1.0f / dsum[(size_t)b * LSEQ + m];
        float* o = Out + ((size_t)b * LSEQ + m) * DMODEL;
        o[d0 + n]      = acc0[i] * inv;
        o[d0 + 64 + n] = acc1[i] * inv;
    }
}

// ---------------------------------------------------------------------------
extern "C" void kernel_launch(void* const* d_in, const int* in_sizes, int n_in,
                              void* d_out, int out_size, void* d_ws, size_t ws_size,
                              hipStream_t stream) {
    (void)in_sizes; (void)n_in; (void)out_size; (void)ws_size;
    const float* A   = (const float*)d_in[0];
    const float* B   = (const float*)d_in[1];
    const float* W_A = (const float*)d_in[2];
    const float* b_A = (const float*)d_in[3];
    const float* W_B = (const float*)d_in[4];
    const float* b_B = (const float*)d_in[5];

    constexpr size_t N_AB   = (size_t)BSZ * LSEQ * DMODEL;
    constexpr size_t N_W    = (size_t)DMODEL * HDIM;
    constexpr size_t N_PROJ = (size_t)BSZ * LSEQ * HDIM;
    constexpr size_t N_WT   = (size_t)BSZ * LSEQ * LSEQ;
    constexpr size_t N_ROW  = (size_t)BSZ * LSEQ;

    char* w = (char*)d_ws;
    size_t off = 0;
    bf16* At   = (bf16*)(w + off); off += N_AB * sizeof(bf16);
    bf16* Bt   = (bf16*)(w + off); off += N_AB * sizeof(bf16);
    bf16* Wat  = (bf16*)(w + off); off += N_W * sizeof(bf16);
    bf16* Wbt  = (bf16*)(w + off); off += N_W * sizeof(bf16);
    bf16* Apj  = (bf16*)(w + off); off += N_PROJ * sizeof(bf16);
    bf16* Bpj  = (bf16*)(w + off); off += N_PROJ * sizeof(bf16);
    bf16* WA   = (bf16*)(w + off); off += N_WT * sizeof(bf16);
    bf16* WBt  = (bf16*)(w + off); off += N_WT * sizeof(bf16);
    float* rmx = (float*)(w + off); off += N_ROW * sizeof(float);
    float* cmx = (float*)(w + off); off += N_ROW * sizeof(float);
    float* rsm = (float*)(w + off); off += N_ROW * sizeof(float);
    float* csm = (float*)(w + off); off += N_ROW * sizeof(float);

    float* A_star = (float*)d_out;
    float* B_star = (float*)d_out + N_AB;

    {
        dim3 grid(DMODEL / 32, LSEQ / 32, BSZ);
        transpose_cast_kernel<<<grid, 256, 0, stream>>>(A, At);
        transpose_cast_kernel<<<grid, 256, 0, stream>>>(B, Bt);
    }
    {
        int n = (int)N_W, blk = 256;
        wcast_kernel<<<(n + blk - 1) / blk, blk, 0, stream>>>(W_A, Wat);
        wcast_kernel<<<(n + blk - 1) / blk, blk, 0, stream>>>(W_B, Wbt);
    }
    {
        dim3 grid(LSEQ / 16, 1, BSZ);
        proj_kernel<<<grid, 128, 0, stream>>>(A, Wat, b_A, Apj);
        proj_kernel<<<grid, 128, 0, stream>>>(B, Wbt, b_B, Bpj);
    }
    {
        int n = (int)(2 * N_ROW);   // rmx and cmx are adjacent
        init_neg_inf_kernel<<<(n + 255) / 256, 256, 0, stream>>>(rmx, n);
        dim3 grid(LSEQ / 128, LSEQ / 16, BSZ);
        scores_max_kernel<<<grid, 256, 0, stream>>>(Apj, Bpj, rmx, cmx);
    }
    {
        dim3 grid(LSEQ / 128, LSEQ / 16, BSZ);
        scores_weights_kernel<<<grid, 256, 0, stream>>>(Apj, Bpj, rmx, cmx, WA, WBt);
    }
    {
        dim3 grid(LSEQ, 1, BSZ);
        rowsum_kernel<<<grid, 256, 0, stream>>>(WA, rsm);
        rowsum_kernel<<<grid, 256, 0, stream>>>(WBt, csm);
    }
    {
        dim3 grid(DMODEL / 128, LSEQ / 16, BSZ);
        weighted_sum_kernel<<<grid, 128, 0, stream>>>(WA, Bt, rsm, A_star);
        weighted_sum_kernel<<<grid, 128, 0, stream>>>(WBt, At, csm, B_star);
    }
}
